// TimeSeriesTransformer_67473936220556
// MI455X (gfx1250) — matrix operand order repaired and checked by reading, not compiled, not run on hardware
//
#include <hip/hip_runtime.h>
#include <hip/hip_bf16.h>

// ---------------------------------------------------------------------------
// TimeSeriesTransformer forward for MI455X (gfx1250, wave32, WMMA bf16).
// Pipeline: proj (q,k,v + vT) -> attention (scores/softmax/a*v) -> 3x MLP.
// All GEMMs use v_wmma_f32_16x16x32_bf16 with f32 accumulation.
// Attention q-slabs are staged to LDS with global_load_async_to_lds_b128
// (ASYNCcnt path); MLP weight slabs are double-buffered in LDS.
// ---------------------------------------------------------------------------

typedef __bf16 bf16_t;
typedef __attribute__((ext_vector_type(16))) __bf16 v16bf;
typedef __attribute__((ext_vector_type(8)))  __bf16 v8bf;
typedef __attribute__((ext_vector_type(2)))  __bf16 v2bf;
typedef __attribute__((ext_vector_type(8)))  float  v8f;
typedef __attribute__((ext_vector_type(4)))  float  v4f;

#define B_DIM 128
#define T_DIM 256
#define N_DIM 128
#define K_DIM 128
#define V_DIM 128
#define F_DIM 2048
#define O_DIM (32 * 128)   // 4096

// ------------------------- WMMA helpers ------------------------------------

__device__ __forceinline__ v8f wmma_bf16(v16bf a, v16bf b, v8f c) {
  // (neg_a, A, neg_b, B, c_mod, C, reuse_a, reuse_b)
  return __builtin_amdgcn_wmma_f32_16x16x32_bf16(false, a, false, b,
                                                 (short)0, c, false, false);
}

// A fragment (16x32, bf16) from a row-major [16+, ld] bf16 matrix.
// ISA layout: lane<16 -> M=lane, K elems {0..7, 16..23}; lane>=16 -> M=lane-16,
// K elems {8..15, 24..31}.  Both segments are 16B-contiguous loads.
__device__ __forceinline__ v16bf load_a_bf16(const bf16_t* base, int ld, int kstep) {
  const int lane = threadIdx.x & 31;
  const int row  = lane & 15;
  const int kb   = (lane & 16) ? 8 : 0;
  const bf16_t* p = base + row * ld + kstep * 32 + kb;
  v8bf lo = *(const v8bf*)(p);
  v8bf hi = *(const v8bf*)(p + 16);
  v16bf a;
#pragma unroll
  for (int i = 0; i < 8; ++i) { a[i] = lo[i]; a[i + 8] = hi[i]; }
  return a;
}

// Same A fragment but sourced from row-major f32 (converted to bf16 in regs).
__device__ __forceinline__ v16bf load_a_f32(const float* base, int ld, int kstep) {
  const int lane = threadIdx.x & 31;
  const int row  = lane & 15;
  const int kb   = (lane & 16) ? 8 : 0;
  const float* p = base + row * ld + kstep * 32 + kb;
  v4f a0 = *(const v4f*)(p);
  v4f a1 = *(const v4f*)(p + 4);
  v4f b0 = *(const v4f*)(p + 16);
  v4f b1 = *(const v4f*)(p + 20);
  v16bf a;
#pragma unroll
  for (int i = 0; i < 4; ++i) {
    a[i]      = (bf16_t)a0[i];
    a[i + 4]  = (bf16_t)a1[i];
    a[i + 8]  = (bf16_t)b0[i];
    a[i + 12] = (bf16_t)b1[i];
  }
  return a;
}

// B fragment (32x16, bf16) from a K-contiguous source: row index = output
// column N, columns = K (i.e. W^T / q / vT row-major layout).
// ISA layout: lane<16 -> N=lane, K {0..7, 16..23}; lane>=16 -> N=lane-16, K {8..15, 24..31}.
__device__ __forceinline__ v16bf load_b_bf16(const bf16_t* base, int ld, int kstep) {
  const int lane = threadIdx.x & 31;
  const int n    = lane & 15;
  const int ks   = kstep * 32 + ((lane & 16) ? 8 : 0);
  const bf16_t* p = base + n * ld + ks;
  v8bf lo = *(const v8bf*)(p);
  v8bf hi = *(const v8bf*)(p + 16);
  v16bf b;
#pragma unroll
  for (int i = 0; i < 8; ++i) { b[i] = lo[i]; b[i + 8] = hi[i]; }
  return b;
}

// Stage a [32 x 16] f32 slab of row-major W into LDS, transposed to
// K-contiguous bf16: wT[n][k], row pitch 32 (64B, keeps ds reads aligned).
// Each thread owns a fixed column n and a contiguous K-chunk: vector b32
// global loads, v_cvt_pk_bf16_f32 packing, single aligned ds_store_b32 each.
template <int NT>
__device__ __forceinline__ void stage_w(const float* W, int ldw, int k0, int n0,
                                        bf16_t* wT) {
  constexpr int KPT = 512 / NT;            // k elements per thread (2 or 4)
  const int t  = threadIdx.x;
  const int n  = t & 15;
  const int kc = (t >> 4) * KPT;
  const float* src = W + (size_t)(k0 + kc) * ldw + (n0 + n);
#pragma unroll
  for (int i = 0; i < KPT; i += 2) {
    const float w0 = src[(size_t)(i)     * ldw];
    const float w1 = src[(size_t)(i + 1) * ldw];
    v2bf p; p[0] = (bf16_t)w0; p[1] = (bf16_t)w1;
    *(v2bf*)(wT + n * 32 + kc + i) = p;
  }
}

// ------------------------- Kernel 1: QKV projection -------------------------
// q = X@Qw+Qb, k = X@Kw+Kb (row-major bf16 [B][T][128]);
// v = X@Vw+Vb stored transposed: vT[B][V][T] bf16 (for K-contiguous B frags).
__global__ __launch_bounds__(128)
void tst_proj_kernel(const float* __restrict__ X,
                     const float* __restrict__ Qw, const float* __restrict__ Qb,
                     const float* __restrict__ Kw, const float* __restrict__ Kb,
                     const float* __restrict__ Vw, const float* __restrict__ Vb,
                     bf16_t* __restrict__ qbf, bf16_t* __restrict__ kbf,
                     bf16_t* __restrict__ vT) {
  __shared__ __align__(16) bf16_t ldsWT[16 * 32];

  int bid   = blockIdx.x;
  const int nTile  = bid & 7;   bid >>= 3;   // 8 column tiles (128/16)
  const int tGroup = bid & 3;   bid >>= 2;   // 4 groups of 4 row tiles
  const int head   = bid % 3;
  const int b      = bid / 3;

  const int wave  = threadIdx.x >> 5;
  const int t0    = (tGroup * 4 + wave) * 16;

  const float* W    = (head == 0) ? Qw : (head == 1) ? Kw : Vw;
  const float* bias = (head == 0) ? Qb : (head == 1) ? Kb : Vb;

  const float* xrow = X + ((size_t)b * T_DIM + t0) * N_DIM;

  v8f acc = {};
#pragma unroll
  for (int ks = 0; ks < 4; ++ks) {           // K = 128 -> 4 steps of 32
    __syncthreads();
    stage_w<128>(W, K_DIM, ks * 32, nTile * 16, ldsWT);
    __syncthreads();
    v16bf a = load_a_f32(xrow, N_DIM, ks);
    v16bf bW = load_b_bf16(ldsWT, 32, 0);
    acc = wmma_bf16(a, bW, acc);
  }

  const int lane = threadIdx.x & 31;
  const int n    = nTile * 16 + (lane & 15);
  const int moff = (lane & 16) ? 8 : 0;
  const float bb = bias[n];

  if (head < 2) {
    bf16_t* dst = ((head == 0) ? qbf : kbf) + (size_t)b * T_DIM * K_DIM;
#pragma unroll
    for (int j = 0; j < 8; ++j)
      dst[(size_t)(t0 + j + moff) * K_DIM + n] = (bf16_t)(acc[j] + bb);
  } else {
    bf16_t* dst = vT + (size_t)b * V_DIM * T_DIM;
#pragma unroll
    for (int j = 0; j < 8; ++j)
      dst[(size_t)n * T_DIM + (t0 + j + moff)] = (bf16_t)(acc[j] + bb);
  }
}

// ------------------------- Kernel 2: attention ------------------------------
// scores[t,s] = k[t,:].q[s,:]; softmax over s; r = a @ v.  One wave owns a
// 16-row tile of t; q s-slabs are DMA'd to LDS (async-to-LDS, ASYNCcnt);
// scores/probs live in LDS.  r stored bf16 [B][T][V].
__global__ __launch_bounds__(64)
void tst_attn_kernel(const bf16_t* __restrict__ qbf,
                     const bf16_t* __restrict__ kbf,
                     const bf16_t* __restrict__ vT,
                     bf16_t* __restrict__ rbf) {
  __shared__ __align__(16) float  s_sc[2][16 * T_DIM];   // 32 KB
  __shared__ __align__(16) bf16_t s_p [2][16 * T_DIM];   // 16 KB
  __shared__ __align__(16) bf16_t s_q [16 * K_DIM];      //  4 KB (shared slab)

  const int b    = blockIdx.x >> 3;
  const int pair = blockIdx.x & 7;
  const int wave = threadIdx.x >> 5;
  const int lane = threadIdx.x & 31;
  const int t0   = (pair * 2 + wave) * 16;

  const bf16_t* kb = kbf + (size_t)b * T_DIM * K_DIM;
  const bf16_t* qb = qbf + (size_t)b * T_DIM * K_DIM;
  const bf16_t* vb = vT  + (size_t)b * V_DIM * T_DIM;

  // Preload the k-tile A fragments (reused for all 16 s-tiles).
  v16bf ak[4];
#pragma unroll
  for (int ks = 0; ks < 4; ++ks) ak[ks] = load_a_bf16(kb + t0 * K_DIM, K_DIM, ks);

  const uint32_t sq_lds = (uint32_t)(uintptr_t)(&s_q[0]);   // LDS byte offset
  float* sc = s_sc[wave];

  for (int sTile = 0; sTile < 16; ++sTile) {
    __syncthreads();   // protect s_q reuse across iterations
    // Async DMA: 16x128 bf16 q-slab (4KB) -> LDS, 16B per lane per issue.
    {
      const char* src = (const char*)(qb + sTile * 16 * K_DIM);
#pragma unroll
      for (int i = 0; i < 4; ++i) {
        const int      idx  = threadIdx.x + i * 64;      // 0..255 x 16B
        const uint32_t ldst = sq_lds + idx * 16;
        const uint64_t gsrc = (uint64_t)(uintptr_t)(src + idx * 16);
        asm volatile("global_load_async_to_lds_b128 %0, %1, off"
                     :: "v"(ldst), "v"(gsrc) : "memory");
      }
      asm volatile("s_wait_asynccnt 0x0" ::: "memory");
    }
    __syncthreads();

    v8f acc = {};
#pragma unroll
    for (int ks = 0; ks < 4; ++ks) {
      v16bf bq = load_b_bf16(s_q, K_DIM, ks);
      acc = wmma_bf16(ak[ks], bq, acc);
    }
    const int n    = sTile * 16 + (lane & 15);
    const int moff = (lane & 16) ? 8 : 0;
#pragma unroll
    for (int j = 0; j < 8; ++j) sc[(j + moff) * T_DIM + n] = acc[j];
  }
  __syncthreads();

  // Softmax over s (256): each row handled by 2 lanes (128 cols each).
  {
    const int r    = lane & 15;
    const int half = lane >> 4;
    const float* row = sc + r * T_DIM + half * 128;
    float m = -3.402823466e+38f;
    for (int i = 0; i < 128; ++i) m = fmaxf(m, row[i]);
    m = fmaxf(m, __shfl_xor(m, 16));
    float s = 0.f;
    for (int i = 0; i < 128; ++i) s += __expf(row[i] - m);
    s += __shfl_xor(s, 16);
    const float inv = 1.f / s;
    bf16_t* pr = s_p[wave] + r * T_DIM + half * 128;
    for (int i = 0; i < 128; ++i) pr[i] = (bf16_t)(__expf(row[i] - m) * inv);
  }
  __syncthreads();

  // r = a(16x256) @ v(256x128): A from LDS probs, B from vT (K-contiguous).
  v16bf ap[8];
#pragma unroll
  for (int ks = 0; ks < 8; ++ks) ap[ks] = load_a_bf16(s_p[wave], T_DIM, ks);

  for (int nTile = 0; nTile < 8; ++nTile) {
    v8f acc = {};
#pragma unroll
    for (int ks = 0; ks < 8; ++ks) {
      v16bf bv = load_b_bf16(vb + nTile * 16 * T_DIM, T_DIM, ks);
      acc = wmma_bf16(ap[ks], bv, acc);
    }
    const int n    = nTile * 16 + (lane & 15);
    const int moff = (lane & 16) ? 8 : 0;
#pragma unroll
    for (int j = 0; j < 8; ++j)
      rbf[((size_t)b * T_DIM + t0 + j + moff) * V_DIM + n] = (bf16_t)acc[j];
  }
}

// ------------------------- Kernel 3: MLP layers -----------------------------
// out[128, N] = act(A[128, K] @ W[K, N] + bias).  One block per 16-col stripe;
// 8 waves cover the 8 row tiles of the 128 batch rows.  W slabs are
// double-buffered in LDS (one barrier per k-step, loads overlap WMMA); the
// slab after next is prefetched into L2 (global_prefetch_b8).
template <bool SIGMOID, typename OutT>
__global__ __launch_bounds__(256)
void tst_mlp_kernel(const bf16_t* __restrict__ A,
                    const float*  __restrict__ W,
                    const float*  __restrict__ bias,
                    OutT* __restrict__ out, int K, int N) {
  __shared__ __align__(16) bf16_t ldsWT[2][16 * 32];

  const int nTile = blockIdx.x;
  const int wave  = threadIdx.x >> 5;
  const int lane  = threadIdx.x & 31;
  const bf16_t* arow = A + (size_t)(wave * 16) * K;

  stage_w<256>(W, N, 0, nTile * 16, ldsWT[0]);

  v8f acc = {};
  int cur = 0;
  for (int k0 = 0; k0 < K; k0 += 32) {
    __syncthreads();                       // stage(cur) visible; buf reuse safe
    if (k0 + 32 < K) {
      stage_w<256>(W, N, k0 + 32, nTile * 16, ldsWT[cur ^ 1]);
      if (k0 + 64 < K)                     // prefetch slab-after-next into L2
        __builtin_prefetch(W + (size_t)(k0 + 64 + lane) * N + nTile * 16, 0, 1);
    }
    v16bf a  = load_a_bf16(arow, K, k0 >> 5);
    v16bf bW = load_b_bf16(ldsWT[cur], 32, 0);
    acc = wmma_bf16(a, bW, acc);
    cur ^= 1;
  }

  const int n    = nTile * 16 + (lane & 15);
  const int moff = (lane & 16) ? 8 : 0;
  const float bb = bias[n];
#pragma unroll
  for (int j = 0; j < 8; ++j) {
    float x = acc[j] + bb;
    if (SIGMOID) x = 1.f / (1.f + __expf(-x));
    out[(size_t)(wave * 16 + j + moff) * N + n] = (OutT)x;
  }
}

// ------------------------- Host-side launcher -------------------------------

extern "C" void kernel_launch(void* const* d_in, const int* in_sizes, int n_in,
                              void* d_out, int out_size, void* d_ws, size_t ws_size,
                              hipStream_t stream) {
  const float* X  = (const float*)d_in[0];
  const float* Qw = (const float*)d_in[1];
  const float* Qb = (const float*)d_in[2];
  const float* Kw = (const float*)d_in[3];
  const float* Kb = (const float*)d_in[4];
  const float* Vw = (const float*)d_in[5];
  const float* Vb = (const float*)d_in[6];
  const float* W1 = (const float*)d_in[7];
  const float* b1 = (const float*)d_in[8];
  const float* W2 = (const float*)d_in[9];
  const float* b2 = (const float*)d_in[10];
  const float* W3 = (const float*)d_in[11];
  const float* b3 = (const float*)d_in[12];
  float* out = (float*)d_out;

  // Workspace layout (bf16): q, k, vT, r, h1, h2  (~33 MB total)
  const size_t SZ_TK = (size_t)B_DIM * T_DIM * K_DIM * sizeof(bf16_t); // 8 MB
  char* ws = (char*)d_ws;
  bf16_t* qbf = (bf16_t*)(ws);
  bf16_t* kbf = (bf16_t*)(ws + SZ_TK);
  bf16_t* vTb = (bf16_t*)(ws + 2 * SZ_TK);
  bf16_t* rbf = (bf16_t*)(ws + 3 * SZ_TK);
  bf16_t* h1  = (bf16_t*)(ws + 4 * SZ_TK);
  bf16_t* h2  = (bf16_t*)(ws + 4 * SZ_TK + (size_t)B_DIM * F_DIM * sizeof(bf16_t));

  // 1) QKV projections: B * 3 heads * 4 tGroups * 8 nTiles blocks of 4 waves.
  tst_proj_kernel<<<B_DIM * 3 * 4 * 8, 128, 0, stream>>>(
      X, Qw, Qb, Kw, Kb, Vw, Vb, qbf, kbf, vTb);

  // 2) Attention: B * 8 blocks of 2 waves (each wave = one 16-row t-tile).
  tst_attn_kernel<<<B_DIM * 8, 64, 0, stream>>>(qbf, kbf, vTb, rbf);

  // 3) MLP: h1 = sigmoid(r @ W1 + b1)   [K = 32768, N = 2048]
  tst_mlp_kernel<true, bf16_t><<<F_DIM / 16, 256, 0, stream>>>(
      rbf, W1, b1, h1, T_DIM * V_DIM, F_DIM);

  // 4) MLP: h2 = sigmoid(h1 @ W2 + b2)  [K = 2048, N = 2048]
  tst_mlp_kernel<true, bf16_t><<<F_DIM / 16, 256, 0, stream>>>(
      h1, W2, b2, h2, F_DIM, F_DIM);

  // 5) out = h2 @ W3 + b3               [K = 2048, N = 4096], f32 output
  tst_mlp_kernel<false, float><<<O_DIM / 16, 256, 0, stream>>>(
      h2, W3, b3, out, F_DIM, O_DIM);
}
